// Mixer2dTriUKAN_66417374265858
// MI455X (gfx1250) — compile-verified, matrix operand order checked by reading, not run on hardware
//
#include <hip/hip_runtime.h>
#include <math.h>

typedef _Float16 f16;
typedef __attribute__((ext_vector_type(16))) _Float16 v16h;
typedef __attribute__((ext_vector_type(8)))  _Float16 h8;
typedef __attribute__((ext_vector_type(8)))  float    v8f;

#if __has_builtin(__builtin_amdgcn_s_wait_asynccnt)
#define WAIT_ASYNC(n) __builtin_amdgcn_s_wait_asynccnt(n)
#else
#define WAIT_ASYNC(n) asm volatile("s_wait_asynccnt %0" ::"i"(n) : "memory")
#endif

// ---------------------------------------------------------------------------
// WMMA GEMM:  C(M,N) f32  =  A(M,K) f16 row-major  x  Bw(N,K) f16 row-major^T
// (Bw stored as weights W[o][k], i.e. B in K x N column-major form.)
//
// Block = 256 threads = 8 waves; block tile = 128(M) x 64(N).  All 8 waves
// share the same 64-column B tile, so B is staged once per block through LDS
// with CDNA5 async-to-LDS copies (ASYNCcnt), double-buffered with issue-ahead
// of one K-step.  The per-wave 16x32 A fragment is software-pipelined by one
// K-step on the direct global path (plus cacheline prefetch).  Per step the
// four B fragments are ds-loaded into distinct registers first, then four
// v_wmma_f32_16x16x32_f16 issue back-to-back sharing one A fragment (RA
// operand-reuse hint on the first three).
//
// EPI: 0 = plain store, 1 = + residual (extra = f32[M*N]),
//      2 = erf-GELU(acc + bias[col]) (extra = f32[N]).
// ---------------------------------------------------------------------------
template <int EPI>
__global__ __launch_bounds__(256)
void gemm_f16_wmma(const f16* __restrict__ A, const f16* __restrict__ Bw,
                   float* __restrict__ C, int M, int N, int K,
                   const float* __restrict__ extra)
{
    // B tile stage: 64 cols x 32 K-halves, row padded to 40 halves (80 B) so
    // 16 parallel ds_load_b128 (stride 20 dwords) hit 16 distinct bank groups.
    __shared__ f16 Bs[2][64][40];

    const int lane = threadIdx.x & 31;
    const int wave = threadIdx.x >> 5;
    const int half = lane >> 4;   // lane group: K sub-block select
    const int lrow = lane & 15;   // A row within tile / B column within tile

    const int row0 = (blockIdx.x * 8 + wave) * 16;
    const int col0 = blockIdx.y * 64;

    const v8f vzero = {0.f, 0.f, 0.f, 0.f, 0.f, 0.f, 0.f, 0.f};
    v8f acc[4];
#pragma unroll
    for (int t = 0; t < 4; ++t) acc[t] = vzero;

    // cooperative async copy of one 64x32 B tile: 256 threads x 16 B
    const int cc  = threadIdx.x >> 2;   // column 0..63
    const int seg = threadIdx.x & 3;    // 8-half segment within the 32-K row
    auto issue_copy = [&](int buf, int k0) {
        const f16* src = Bw + (size_t)(col0 + cc) * K + k0 + seg * 8;
        unsigned   dst = (unsigned)(size_t)&Bs[buf][cc][seg * 8];
        asm volatile("global_load_async_to_lds_b128 %0, %1, off"
                     ::"v"(dst), "v"(src) : "memory");
    };

    // A fragment 16x32: lanes 0-15 row m hold K {0..7,16..23}(+k0),
    // lanes 16-31 row m hold K {8..15,24..31}(+k0)  (ISA 7.12.2)
    const f16* arow = A + (size_t)(row0 + lrow) * K + half * 8;
    auto load_a = [&](int k0) -> v16h {
        h8 lo = *(const h8*)(arow + k0);
        h8 hi = *(const h8*)(arow + k0 + 16);
        v16h r;
#pragma unroll
        for (int i = 0; i < 8; ++i) { r[i] = lo[i]; r[i + 8] = hi[i]; }
        return r;
    };

    const int nsteps = K >> 5;
    issue_copy(0, 0);
    if (nsteps > 1) issue_copy(1, 32);

    v16h acur = load_a(0);      // software-pipelined A fragment

    for (int s = 0; s < nsteps; ++s) {
        const int buf = s & 1;
        const int k0  = s << 5;

        if (s + 1 < nsteps) WAIT_ASYNC(1); else WAIT_ASYNC(0);
        __syncthreads();                 // whole B tile resident in Bs[buf]

        if (k0 + 64 < K) __builtin_prefetch(arow + k0 + 64, 0, 1);

        // issue next A fragment loads now; first use is next iteration
        v16h anext = acur;
        if (s + 1 < nsteps) anext = load_a(k0 + 32);

        // all four B fragments into distinct registers -> one dscnt wait,
        // then four back-to-back WMMAs
        v16h bfrag[4];
#pragma unroll
        for (int t = 0; t < 4; ++t) {
            const f16* brow = &Bs[buf][t * 16 + lrow][half * 8];
            h8 lo = *(const h8*)(brow);
            h8 hi = *(const h8*)(brow + 16);
#pragma unroll
            for (int i = 0; i < 8; ++i) { bfrag[t][i] = lo[i]; bfrag[t][i + 8] = hi[i]; }
        }

        acc[0] = __builtin_amdgcn_wmma_f32_16x16x32_f16(
            false, acur, false, bfrag[0], (short)0, acc[0], true,  false);
        acc[1] = __builtin_amdgcn_wmma_f32_16x16x32_f16(
            false, acur, false, bfrag[1], (short)0, acc[1], true,  false);
        acc[2] = __builtin_amdgcn_wmma_f32_16x16x32_f16(
            false, acur, false, bfrag[2], (short)0, acc[2], true,  false);
        acc[3] = __builtin_amdgcn_wmma_f32_16x16x32_f16(
            false, acur, false, bfrag[3], (short)0, acc[3], false, false);

        __syncthreads();                 // all waves done reading Bs[buf]
        if (s + 2 < nsteps) issue_copy(buf, (s + 2) << 5);
        acur = anext;
    }

    // C/D layout: lanes 0-15 -> N=lane, M=vgpr; lanes 16-31 -> N=lane-16, M=8+vgpr
#pragma unroll
    for (int t = 0; t < 4; ++t) {
        const int col = col0 + t * 16 + lrow;
#pragma unroll
        for (int v = 0; v < 8; ++v) {
            const int row = row0 + half * 8 + v;
            const size_t idx = (size_t)row * N + col;
            float r = acc[t][v];
            if (EPI == 1) {
                r += extra[idx];
            } else if (EPI == 2) {
                float u = r + extra[col];
                r = 0.5f * u * (1.0f + erff(u * 0.70710678118654752f));
            }
            C[idx] = r;
        }
    }
}

// ---------------------------------------------------------------------------
// Joint LayerNorm: per batch sample, mean/var over all C*L elements, then
// y[b,c,l] = (x - mu)/sqrt(var+eps) * w[l*C+c] + bias[l*C+c]
// (equivalent to the reference's permute -> LN over (L,C) -> permute back)
// ---------------------------------------------------------------------------
__global__ __launch_bounds__(256)
void ln_joint(const float* __restrict__ X, const float* __restrict__ w,
              const float* __restrict__ bia, float* __restrict__ Y,
              int C, int L)
{
    const int b = blockIdx.x;
    const int n = C * L;
    const float* xb = X + (size_t)b * n;
    float s = 0.f, ss = 0.f;
    for (int i = threadIdx.x; i < n; i += blockDim.x) {
        float v = xb[i];
        s += v; ss += v * v;
    }
    __shared__ float rs[256], rss[256];
    rs[threadIdx.x] = s; rss[threadIdx.x] = ss;
    __syncthreads();
    for (int o = 128; o > 0; o >>= 1) {
        if (threadIdx.x < o) {
            rs[threadIdx.x]  += rs[threadIdx.x + o];
            rss[threadIdx.x] += rss[threadIdx.x + o];
        }
        __syncthreads();
    }
    const float mu  = rs[0] / (float)n;
    const float var = rss[0] / (float)n - mu * mu;
    const float inv = rsqrtf(var + 1e-5f);
    float* yb = Y + (size_t)b * n;
    for (int i = threadIdx.x; i < n; i += blockDim.x) {
        int c = i / L, l = i - c * L;
        yb[i] = (xb[i] - mu) * inv * w[l * C + c] + bia[l * C + c];
    }
}

// ---------------------------------------------------------------------------
// KAN activation prep: for each input element x -> 9 f16 values
//   [silu(x), B0(x)..B7(x)]  (cubic Cox-de Boor on uniform grid, h = 2/5,
//   knot j at (j-3)*h - 1; denominators collapse to k*h)
// Aug layout: (Nrows, I*9), augmented K-index = i*9 + j
// ---------------------------------------------------------------------------
__global__ __launch_bounds__(256)
void kan_prep(const float* __restrict__ X, f16* __restrict__ Aug,
              int total, int I)
{
    int idx = blockIdx.x * blockDim.x + threadIdx.x;
    if (idx >= total) return;
    const int n = idx / I, i = idx - n * I;
    const float x = X[idx];
    const float silu = x / (1.0f + __expf(-x));

    const float h = 0.4f;
    float b[11];
#pragma unroll
    for (int j = 0; j < 11; ++j) {
        float gj = (float)(j - 3) * h - 1.0f;
        b[j] = (x >= gj && x < gj + h) ? 1.0f : 0.0f;
    }
#pragma unroll
    for (int k = 1; k <= 3; ++k) {
        const float inv = 1.0f / ((float)k * h);
#pragma unroll
        for (int j = 0; j < 11 - k; ++j) {
            float gj   = (float)(j - 3) * h - 1.0f;
            float gjk1 = (float)(j + k - 2) * h - 1.0f;  // g[j+k+1]
            b[j] = ((x - gj) * b[j] + (gjk1 - x) * b[j + 1]) * inv;
        }
    }
    f16* o = Aug + (size_t)n * (size_t)(I * 9) + (size_t)i * 9;
    o[0] = (f16)silu;
#pragma unroll
    for (int k = 0; k < 8; ++k) o[1 + k] = (f16)b[k];
}

// Pack KAN weights: W16[o][i*9+0] = base[o][i]; W16[o][i*9+1+k] = spline[o][i][k]
__global__ __launch_bounds__(256)
void kan_wprep(const float* __restrict__ base, const float* __restrict__ spline,
               f16* __restrict__ W, int O, int I)
{
    int idx = blockIdx.x * blockDim.x + threadIdx.x;
    if (idx >= O * I) return;
    const int o = idx / I, i = idx - o * I;
    f16* w = W + (size_t)o * (size_t)(I * 9) + (size_t)i * 9;
    w[0] = (f16)base[idx];
    const float* sp = spline + (size_t)idx * 8;
#pragma unroll
    for (int k = 0; k < 8; ++k) w[1 + k] = (f16)sp[k];
}

// softmax rows sum to 1 => gcn concat([x, x*1, x*1]) @ W^T folds to one matrix:
// Weff[o][k] = W[o][k] + W[o][L+k] + W[o][2L+k]
__global__ __launch_bounds__(256)
void fold_gcn_w(const float* __restrict__ w3, f16* __restrict__ W, int L)
{
    int idx = blockIdx.x * blockDim.x + threadIdx.x;
    if (idx >= L * L) return;
    const int o = idx / L, k = idx - o * L;
    const float* r = w3 + (size_t)o * 3 * L;
    W[(size_t)o * L + k] = (f16)(r[k] + r[L + k] + r[2 * L + k]);
}

__global__ __launch_bounds__(256)
void cvt_f32_f16(const float* __restrict__ X, f16* __restrict__ Y, int n)
{
    int idx = blockIdx.x * blockDim.x + threadIdx.x;
    if (idx < n) Y[idx] = (f16)X[idx];
}

// ---------------------------------------------------------------------------
// Orchestration
// ---------------------------------------------------------------------------
extern "C" void kernel_launch(void* const* d_in, const int* in_sizes, int n_in,
                              void* d_out, int out_size, void* d_ws, size_t ws_size,
                              hipStream_t stream)
{
    (void)in_sizes; (void)n_in; (void)out_size; (void)ws_size;

    const int B = 16, C4 = 128, T = 512, D = 512, H = 64;
    const int NR = B * C4;          // 2048 GEMM rows
    const int KT = T * 9;           // 4608
    const int KH = H * 9;           // 576

    const float* x            = (const float*)d_in[0];
    const float* tm1_ln_w     = (const float*)d_in[1];
    const float* tm1_ln_b     = (const float*)d_in[2];
    const float* tm1_k1_base  = (const float*)d_in[3];
    const float* tm1_k1_spl   = (const float*)d_in[4];
    const float* tm1_k2_base  = (const float*)d_in[5];
    const float* tm1_k2_spl   = (const float*)d_in[6];
    /* d_in[7] = A1: mathematically dead (softmax rows sum to 1) */
    const float* g1_w         = (const float*)d_in[8];
    const float* g1_b         = (const float*)d_in[9];
    const float* k1_base      = (const float*)d_in[10];
    const float* k1_spl       = (const float*)d_in[11];
    const float* tm_ln_w      = (const float*)d_in[12];
    const float* tm_ln_b      = (const float*)d_in[13];
    const float* tm_k1_base   = (const float*)d_in[14];
    const float* tm_k1_spl    = (const float*)d_in[15];
    const float* tm_k2_base   = (const float*)d_in[16];
    const float* tm_k2_spl    = (const float*)d_in[17];
    /* d_in[18] = A2: dead */
    const float* g2_w         = (const float*)d_in[19];
    const float* g2_b         = (const float*)d_in[20];
    const float* k2_base      = (const float*)d_in[21];
    const float* k2_spl       = (const float*)d_in[22];

    // workspace carve-up
    char* ws = (char*)d_ws;
    size_t off = 0;
    auto carve = [&](size_t bytes) -> void* {
        off = (off + 255) & ~(size_t)255;
        void* p = ws + off;
        off += bytes;
        return p;
    };
    f16*   aug   = (f16*)  carve((size_t)NR * KT * sizeof(f16));   // 18.9 MB
    f16*   wbig  = (f16*)  carve((size_t)512 * KT * sizeof(f16));  //  4.7 MB
    f16*   a16   = (f16*)  carve((size_t)NR * 512 * sizeof(f16));  //  2.0 MB
    f16*   weff  = (f16*)  carve((size_t)512 * 512 * sizeof(f16)); //  0.5 MB
    float* zln   = (float*)carve((size_t)NR * 512 * sizeof(float));
    float* h1    = (float*)carve((size_t)NR * 64  * sizeof(float));
    float* tm1   = (float*)carve((size_t)NR * 512 * sizeof(float));
    float* y1    = (float*)carve((size_t)NR * 512 * sizeof(float));
    float* cm    = (float*)carve((size_t)NR * 512 * sizeof(float));
    float* tmb   = (float*)carve((size_t)NR * 512 * sizeof(float));
    float* y2    = (float*)carve((size_t)NR * 512 * sizeof(float));
    float* outp  = (float*)d_out;

    const dim3 blk(256);
    auto ceil_g = [](int n) { return dim3((unsigned)((n + 255) / 256)); };
    auto gemm_grid = [&](int M, int N) { return dim3((unsigned)(M / 128), (unsigned)(N / 64)); };

    // ---- Stage A: tm1 = token_mixing(x) -------------------------------------
    ln_joint<<<dim3(B), blk, 0, stream>>>(x, tm1_ln_w, tm1_ln_b, zln, C4, T);
    kan_prep<<<ceil_g(NR * T), blk, 0, stream>>>(zln, aug, NR * T, T);
    kan_wprep<<<ceil_g(H * T), blk, 0, stream>>>(tm1_k1_base, tm1_k1_spl, wbig, H, T);
    gemm_f16_wmma<0><<<gemm_grid(NR, H), blk, 0, stream>>>(aug, wbig, h1, NR, H, KT, nullptr);
    kan_prep<<<ceil_g(NR * H), blk, 0, stream>>>(h1, aug, NR * H, H);
    kan_wprep<<<ceil_g(T * H), blk, 0, stream>>>(tm1_k2_base, tm1_k2_spl, wbig, T, H);
    gemm_f16_wmma<1><<<gemm_grid(NR, T), blk, 0, stream>>>(aug, wbig, tm1, NR, T, KH, x);

    // ---- Stage B: y1 = gelu(tm1 @ fold(g1_w)^T + g1_b) ----------------------
    fold_gcn_w<<<ceil_g(T * T), blk, 0, stream>>>(g1_w, weff, T);
    cvt_f32_f16<<<ceil_g(NR * T), blk, 0, stream>>>(tm1, a16, NR * T);
    gemm_f16_wmma<2><<<gemm_grid(NR, T), blk, 0, stream>>>(a16, weff, y1, NR, T, T, g1_b);

    // ---- Stage C: cm = kan(x, k1) -------------------------------------------
    kan_prep<<<ceil_g(NR * T), blk, 0, stream>>>(x, aug, NR * T, T);
    kan_wprep<<<ceil_g(D * T), blk, 0, stream>>>(k1_base, k1_spl, wbig, D, T);
    gemm_f16_wmma<0><<<gemm_grid(NR, D), blk, 0, stream>>>(aug, wbig, cm, NR, D, KT, nullptr);

    // ---- Stage D: tm = token_mixing(cm) -------------------------------------
    ln_joint<<<dim3(B), blk, 0, stream>>>(cm, tm_ln_w, tm_ln_b, zln, C4, D);
    kan_prep<<<ceil_g(NR * D), blk, 0, stream>>>(zln, aug, NR * D, D);
    kan_wprep<<<ceil_g(H * D), blk, 0, stream>>>(tm_k1_base, tm_k1_spl, wbig, H, D);
    gemm_f16_wmma<0><<<gemm_grid(NR, H), blk, 0, stream>>>(aug, wbig, h1, NR, H, KT, nullptr);
    kan_prep<<<ceil_g(NR * H), blk, 0, stream>>>(h1, aug, NR * H, H);
    kan_wprep<<<ceil_g(D * H), blk, 0, stream>>>(tm_k2_base, tm_k2_spl, wbig, D, H);
    gemm_f16_wmma<1><<<gemm_grid(NR, D), blk, 0, stream>>>(aug, wbig, tmb, NR, D, KH, cm);

    // ---- Stage E: y2 = gelu(tm @ fold(g2_w)^T + g2_b) -----------------------
    fold_gcn_w<<<ceil_g(D * D), blk, 0, stream>>>(g2_w, weff, D);
    cvt_f32_f16<<<ceil_g(NR * D), blk, 0, stream>>>(tmb, a16, NR * D);
    gemm_f16_wmma<2><<<gemm_grid(NR, D), blk, 0, stream>>>(a16, weff, y2, NR, D, D, g2_b);

    // ---- Stage F: out = y1 + kan(y2, k2) ------------------------------------
    kan_prep<<<ceil_g(NR * D), blk, 0, stream>>>(y2, aug, NR * D, D);
    kan_wprep<<<ceil_g(T * D), blk, 0, stream>>>(k2_base, k2_spl, wbig, T, D);
    gemm_f16_wmma<1><<<gemm_grid(NR, T), blk, 0, stream>>>(aug, wbig, outp, NR, T, (D * 9), y1);
}
